// BaseGINE_32908039422399
// MI455X (gfx1250) — matrix operand order repaired and checked by reading, not compiled, not run on hardware
//
#include <hip/hip_runtime.h>
#include <hip/hip_bf16.h>
#include <math.h>

// ---------------------------------------------------------------------------
// GINE (3-layer) for MI455X / gfx1250, wave32, WMMA f32_16x16x32_f16.
// D = 64 fixed. Memory-bound; GEMMs via v_wmma, scatter via f32 atomics.
// Round 2: combine/GELU modes are template params -> branch-free GEMM body.
// ---------------------------------------------------------------------------

typedef __attribute__((ext_vector_type(16))) _Float16 v16h;
typedef __attribute__((ext_vector_type(8)))  float    v8f;

#define DD 64
#define INV_SQRT2_F 0.70710678118654752f

__device__ __forceinline__ float gelu_erf(float v) {
    return 0.5f * v * (1.0f + erff(v * INV_SQRT2_F));
}

// ---------------------------------------------------------------------------
// Pack a 64x64 row-major fp32 weight matrix into per-lane B fragments (f16).
// Fragment layout (dense 16x16x32 f16 B, 32K x 16N, wave32):
//   lane holds column n = lane & 15
//   K chunk for lane half h = lane>>4, vgpr v:  k = kt*32 + (v>=4)*16 + h*8 + (v&3)*2 (+0/1)
// Flat index: (((kt*4 + nt)*32 + lane)*8 + v)   -> one uint (half2) each.
// Each lane's 8 half2 are contiguous => a single 32B v16h load in the GEMM.
// Matrices: id 0 = W_be, 1..L = W1[l], L+1..2L = W2[l].
// ---------------------------------------------------------------------------
__global__ void pack_weights_kernel(const float* __restrict__ W_be,
                                    const float* __restrict__ W1,
                                    const float* __restrict__ W2,
                                    unsigned int* __restrict__ Bp, int L) {
    int mId = blockIdx.x;
    const float* W;
    if (mId == 0)       W = W_be;
    else if (mId <= L)  W = W1 + (size_t)(mId - 1) * DD * DD;
    else                W = W2 + (size_t)(mId - 1 - L) * DD * DD;
    unsigned int* dst = Bp + (size_t)mId * 2048;

    for (int idx = threadIdx.x; idx < 2048; idx += blockDim.x) {
        int v    = idx & 7;
        int lane = (idx >> 3) & 31;
        int nt   = (idx >> 8) & 3;
        int kt   = idx >> 10;
        int n = nt * 16 + (lane & 15);
        int k = kt * 32 + ((v >= 4) ? 16 : 0) + ((lane >> 4) & 1) * 8 + (v & 3) * 2;
        _Float16 h0 = (_Float16)W[(size_t)k * DD + n];
        _Float16 h1 = (_Float16)W[(size_t)(k + 1) * DD + n];
        unsigned short u0 = __builtin_bit_cast(unsigned short, h0);
        unsigned short u1 = __builtin_bit_cast(unsigned short, h1);
        dst[idx] = (unsigned int)u0 | ((unsigned int)u1 << 16);
    }
}

__global__ void fill_zero_kernel(float* __restrict__ p, int n) {
    int t = blockIdx.x * blockDim.x + threadIdx.x;
    if (t < n) p[t] = 0.0f;
}

// ---------------------------------------------------------------------------
// 64-wide GEMM: out[rows x 64] = op(A) @ W + bias, optional GELU.
// COMBINE=false: op(A) = A
// COMBINE=true : op(A) = (1+eps[layer])*A + Aadd      (GINE combine, fused)
// One wave per 16-row tile; 8 x v_wmma_f32_16x16x32_f16 per tile.
// rows must be a multiple of 16 (E=800000, N=50000 both are).
// ---------------------------------------------------------------------------
template <bool COMBINE, bool APPLY_GELU>
__global__ void gemm64_wmma_kernel(const float* __restrict__ A,
                                   const float* __restrict__ Aadd,
                                   const float* __restrict__ epsArr, int layer,
                                   const unsigned int* __restrict__ Bpack,
                                   const float* __restrict__ bias,
                                   float* __restrict__ out,
                                   int rows) {
    int wave = blockIdx.x * (blockDim.x >> 5) + (threadIdx.x >> 5);
    int lane = threadIdx.x & 31;
    int rowBase = wave * 16;
    if (rowBase >= rows) return;

    float scale = 1.0f;
    if (COMBINE) scale = 1.0f + epsArr[layer];

    int m = rowBase + (lane & 15);
    int half8 = ((lane >> 4) & 1) * 8;              // A-fragment K sub-offset
    const float* rowp  = A + (size_t)m * DD;
    const float* rowp2 = COMBINE ? (Aadd + (size_t)m * DD) : nullptr;

    const v16h* Bmat = reinterpret_cast<const v16h*>(Bpack);

    v8f acc[4] = {};

    #pragma unroll
    for (int kt = 0; kt < 2; ++kt) {
        // ---- build A fragment (ISA 16-bit A 16x32 layout), branch-free ----
        float2 f[8];
        #pragma unroll
        for (int v = 0; v < 8; ++v) {
            int k = kt * 32 + ((v >= 4) ? 16 : 0) + half8 + (v & 3) * 2;
            f[v] = *reinterpret_cast<const float2*>(rowp + k);
        }
        if (COMBINE) {
            float2 g[8];
            #pragma unroll
            for (int v = 0; v < 8; ++v) {
                int k = kt * 32 + ((v >= 4) ? 16 : 0) + half8 + (v & 3) * 2;
                g[v] = *reinterpret_cast<const float2*>(rowp2 + k);
            }
            #pragma unroll
            for (int v = 0; v < 8; ++v) {
                f[v].x = fmaf(scale, f[v].x, g[v].x);
                f[v].y = fmaf(scale, f[v].y, g[v].y);
            }
        }
        v16h a;
        #pragma unroll
        for (int v = 0; v < 8; ++v) {
            a[2 * v]     = (_Float16)f[v].x;
            a[2 * v + 1] = (_Float16)f[v].y;
        }
        // ---- 4 column tiles ----
        #pragma unroll
        for (int nt = 0; nt < 4; ++nt) {
            v16h b = Bmat[(size_t)(kt * 4 + nt) * 32 + lane];
            acc[nt] = __builtin_amdgcn_wmma_f32_16x16x32_f16(
                false, a, false, b, (short)0, acc[nt], false, false);
        }
    }

    // ---- epilogue: bias (+ GELU), store fp32 (C/D layout: m = v + half*8) ----
    int n0   = lane & 15;
    int mofs = ((lane >> 4) & 1) * 8;
    #pragma unroll
    for (int nt = 0; nt < 4; ++nt) {
        int n = nt * 16 + n0;
        float bv = bias[n];
        #pragma unroll
        for (int v = 0; v < 8; ++v) {
            int mr = rowBase + mofs + v;
            float val = acc[nt][v] + bv;
            if (APPLY_GELU) val = gelu_erf(val);
            out[(size_t)mr * DD + n] = val;
        }
    }
}

// ---------------------------------------------------------------------------
// m = GELU(x[src] + edge_emb); agg[dst] += m   (f32 atomics, float4 per thread)
// ---------------------------------------------------------------------------
__global__ void gather_scatter_kernel(const float* __restrict__ x,
                                      const float* __restrict__ edge_emb,
                                      const int* __restrict__ src,
                                      const int* __restrict__ dst,
                                      float* __restrict__ agg, int E_) {
    int t = blockIdx.x * blockDim.x + threadIdx.x;
    if (t >= E_ * 16) return;
    int e  = t >> 4;
    int d4 = (t & 15) * 4;
    int s  = src[e];
    int dn = dst[e];
    float4 xe = *reinterpret_cast<const float4*>(x + (size_t)s * DD + d4);
    float4 ee = *reinterpret_cast<const float4*>(edge_emb + (size_t)e * DD + d4);
    float m0 = gelu_erf(xe.x + ee.x);
    float m1 = gelu_erf(xe.y + ee.y);
    float m2 = gelu_erf(xe.z + ee.z);
    float m3 = gelu_erf(xe.w + ee.w);
    float* ap = agg + (size_t)dn * DD + d4;
    atomicAdd(ap + 0, m0);
    atomicAdd(ap + 1, m1);
    atomicAdd(ap + 2, m2);
    atomicAdd(ap + 3, m3);
}

// ---------------------------------------------------------------------------
// BN stats: stats[0:64] = sum per column, stats[64:128] = sum of squares.
// ---------------------------------------------------------------------------
__global__ void bn_stats_kernel(const float* __restrict__ h,
                                float* __restrict__ stats, int rows) {
    int d      = threadIdx.x & 63;
    int rlocal = threadIdx.x >> 6;                  // 0..3
    int rstride = gridDim.x * 4;
    float s = 0.0f, s2 = 0.0f;
    for (int r = blockIdx.x * 4 + rlocal; r < rows; r += rstride) {
        float v = h[(size_t)r * DD + d];
        s += v; s2 += v * v;
    }
    __shared__ float sh[2][4][DD];
    sh[0][rlocal][d] = s;
    sh[1][rlocal][d] = s2;
    __syncthreads();
    if (rlocal == 0) {
        s  = sh[0][0][d] + sh[0][1][d] + sh[0][2][d] + sh[0][3][d];
        s2 = sh[1][0][d] + sh[1][1][d] + sh[1][2][d] + sh[1][3][d];
        atomicAdd(&stats[d], s);
        atomicAdd(&stats[DD + d], s2);
    }
}

// ---------------------------------------------------------------------------
// x_out = (x_in + GELU(BN(h2))) * 1/sqrt(2)
// ---------------------------------------------------------------------------
__global__ void bn_gelu_residual_kernel(const float* __restrict__ xin,
                                        const float* __restrict__ h2,
                                        const float* __restrict__ stats,
                                        const float* __restrict__ gamma,
                                        const float* __restrict__ beta,
                                        float* __restrict__ xout, int rows) {
    int t = blockIdx.x * blockDim.x + threadIdx.x;
    if (t >= rows * 16) return;
    int r  = t >> 4;
    int d4 = (t & 15) * 4;
    float invN = 1.0f / (float)rows;
    #pragma unroll
    for (int j = 0; j < 4; ++j) {
        int d = d4 + j;
        float mu  = stats[d] * invN;
        float var = stats[DD + d] * invN - mu * mu;
        float rs  = rsqrtf(var + 1e-5f);
        size_t idx = (size_t)r * DD + d;
        float hn = (h2[idx] - mu) * rs * gamma[d] + beta[d];
        xout[idx] = (xin[idx] + gelu_erf(hn)) * INV_SQRT2_F;
    }
}

// ---------------------------------------------------------------------------
// Launcher
// ---------------------------------------------------------------------------
extern "C" void kernel_launch(void* const* d_in, const int* in_sizes, int n_in,
                              void* d_out, int out_size, void* d_ws, size_t ws_size,
                              hipStream_t stream) {
    const float* x_in0 = (const float*)d_in[0];
    const int*   ei    = (const int*)  d_in[1];
    const float* ea    = (const float*)d_in[2];
    const float* W_be  = (const float*)d_in[3];
    const float* b_be  = (const float*)d_in[4];
    const float* eps   = (const float*)d_in[5];
    const float* W1    = (const float*)d_in[6];
    const float* b1    = (const float*)d_in[7];
    const float* W2    = (const float*)d_in[8];
    const float* b2    = (const float*)d_in[9];
    const float* gamma = (const float*)d_in[10];
    const float* beta  = (const float*)d_in[11];

    const int N = in_sizes[0] / DD;
    const int E = in_sizes[1] / 2;
    const int L = in_sizes[5];

    const int* src = ei;
    const int* dst = ei + E;

    // --- workspace carving (256B aligned) ---
    char*  ws  = (char*)d_ws;
    size_t off = 0;
    auto take = [&](size_t bytes) -> char* {
        char* p = ws + off;
        off = (off + bytes + 255) & ~(size_t)255;
        return p;
    };
    float* edge_emb = (float*)take((size_t)E * DD * sizeof(float));
    float* agg      = (float*)take(((size_t)N * DD + 2 * DD) * sizeof(float));
    float* stats    = agg + (size_t)N * DD;          // contiguous after agg
    float* u        = (float*)take((size_t)N * DD * sizeof(float));
    float* h2       = (float*)take((size_t)N * DD * sizeof(float));
    float* xA       = (float*)take((size_t)N * DD * sizeof(float));
    float* xB       = (float*)take((size_t)N * DD * sizeof(float));
    unsigned int* wpack = (unsigned int*)take((size_t)(1 + 2 * L) * 2048 * sizeof(unsigned int));

    // --- pack all 64x64 weights into WMMA B fragments (f16) ---
    pack_weights_kernel<<<1 + 2 * L, 256, 0, stream>>>(W_be, W1, W2, wpack, L);

    // --- edge_emb = edge_attr @ W_be + b_be  (once) ---
    {
        int waves  = E / 16;
        int blocks = (waves + 3) / 4;
        gemm64_wmma_kernel<false, false><<<blocks, 128, 0, stream>>>(
            ea, nullptr, nullptr, 0, wpack, b_be, edge_emb, E);
    }

    int nWaves  = N / 16;
    int nBlocks = (nWaves + 3) / 4;

    for (int l = 0; l < L; ++l) {
        const float* xin  = (l == 0) ? x_in0 : ((l == 1) ? xA : xB);
        float*       xout = (l == L - 1) ? (float*)d_out : ((l == 0) ? xA : xB);

        // zero agg + stats (contiguous)
        {
            int n = N * DD + 2 * DD;
            fill_zero_kernel<<<(n + 255) / 256, 256, 0, stream>>>(agg, n);
        }
        // m = GELU(x[src] + edge_emb); agg[dst] += m
        {
            int n = E * 16;
            gather_scatter_kernel<<<(n + 255) / 256, 256, 0, stream>>>(
                xin, edge_emb, src, dst, agg, E);
        }
        // u = GELU(((1+eps)*x + agg) @ W1[l] + b1[l])
        gemm64_wmma_kernel<true, true><<<nBlocks, 128, 0, stream>>>(
            xin, agg, eps, l, wpack + (size_t)(1 + l) * 2048,
            b1 + (size_t)l * DD, u, N);
        // h2 = u @ W2[l] + b2[l]
        gemm64_wmma_kernel<false, false><<<nBlocks, 128, 0, stream>>>(
            u, nullptr, nullptr, 0, wpack + (size_t)(1 + L + l) * 2048,
            b2 + (size_t)l * DD, h2, N);
        // BN stats
        bn_stats_kernel<<<208, 256, 0, stream>>>(h2, stats, N);
        // x_out = (x_in + GELU(BN(h2))) / sqrt(2)
        {
            int n = N * 16;
            bn_gelu_residual_kernel<<<(n + 255) / 256, 256, 0, stream>>>(
                xin, h2, stats, gamma + (size_t)l * DD, beta + (size_t)l * DD,
                xout, N);
        }
    }
}
